// Model_45500883534033
// MI455X (gfx1250) — matrix operand order repaired and checked by reading, not compiled, not run on hardware
//
#include <hip/hip_runtime.h>
#include <math.h>

// ---------------- problem constants ----------------
#define B_    2048
#define L_    10
#define CIN_  5
#define D_    512
#define H_    8
#define DH_   64
#define DFF_  512
#define M_    (B_ * L_)      // 20480 token rows
#define TOPK_ 6

// ---------------- vector types ----------------
typedef __attribute__((ext_vector_type(4)))  _Float16 half4_t;
typedef __attribute__((ext_vector_type(8)))  _Float16 half8_t;
typedef __attribute__((ext_vector_type(16))) _Float16 half16_t;
typedef __attribute__((ext_vector_type(8)))  float    float8_t;

// =====================================================================
// WMMA GEMM:  C[M,N] = A[M,K] * W[N,K]^T + bias[N]   (optional ReLU)
// M = 20480, N = K = 512 for every GEMM in this model.
// Block: 256 threads (8 waves), tile 128x128, BK=32,
// per wave: 2x4 grid of 16x16 v_wmma_f32_16x16x32_f16 tiles.
// =====================================================================
#define BM  128
#define BN  128
#define BK  32
#define LDT 40   // padded LDS row stride (halves): conflict-free, 16B aligned

__global__ __launch_bounds__(256)
void gemm_wmma_f16(const float* __restrict__ A, const float* __restrict__ W,
                   const float* __restrict__ bias, float* __restrict__ C,
                   int relu)
{
    __shared__ __align__(16) _Float16 As[BM * LDT];
    __shared__ __align__(16) _Float16 Bs[BN * LDT];

    const int K = D_;
    const int N = D_;
    const int tid   = threadIdx.x;
    const int m0    = blockIdx.x * BM;
    const int n0    = blockIdx.y * BN;
    const int wave  = tid >> 5;
    const int lane  = tid & 31;
    const int lidx  = lane & 15;   // row/col within 16
    const int lhalf = lane >> 4;   // which K half-group
    const int m_off = (wave & 3) * 32;   // 4 waves along M
    const int n_off = (wave >> 2) * 64;  // 2 waves along N

    float8_t zero = {0.f, 0.f, 0.f, 0.f, 0.f, 0.f, 0.f, 0.f};
    float8_t acc[2][4];
#pragma unroll
    for (int mi = 0; mi < 2; ++mi)
#pragma unroll
        for (int ni = 0; ni < 4; ++ni) acc[mi][ni] = zero;

    for (int kt = 0; kt < K; kt += BK) {
        // ---- stage A (activations) and B (weights) f32 -> f16 into LDS ----
#pragma unroll
        for (int i = tid; i < (BM * BK) / 4; i += 256) {  // 1024 float4 slots
            const int row = i >> 3;
            const int c4  = (i & 7) * 4;
            float4 fa = *(const float4*)(A + (size_t)(m0 + row) * K + kt + c4);
            half4_t ha = { (_Float16)fa.x, (_Float16)fa.y,
                           (_Float16)fa.z, (_Float16)fa.w };
            *(half4_t*)(As + row * LDT + c4) = ha;
            float4 fb = *(const float4*)(W + (size_t)(n0 + row) * K + kt + c4);
            half4_t hb = { (_Float16)fb.x, (_Float16)fb.y,
                           (_Float16)fb.z, (_Float16)fb.w };
            *(half4_t*)(Bs + row * LDT + c4) = hb;
        }
        // prefetch next k-tile (global_prefetch_b8)
        if (kt + BK < K) {
            const int prow = tid >> 1;
            const int pcol = (tid & 1) * 16;
            __builtin_prefetch(A + (size_t)(m0 + prow) * K + kt + BK + pcol, 0, 0);
            __builtin_prefetch(W + (size_t)(n0 + prow) * K + kt + BK + pcol, 0, 0);
        }
        __syncthreads();

        // ---- load fragments per ISA layouts ----
        half16_t af[2];
#pragma unroll
        for (int mi = 0; mi < 2; ++mi) {
            const _Float16* p = As + (m_off + mi * 16 + lidx) * LDT + lhalf * 8;
            half8_t lo = *(const half8_t*)p;          // K = koff .. koff+7
            half8_t hi = *(const half8_t*)(p + 16);   // K = 16+koff .. 23+koff
            af[mi] = __builtin_shufflevector(lo, hi,
                     0, 1, 2, 3, 4, 5, 6, 7, 8, 9, 10, 11, 12, 13, 14, 15);
        }
        half16_t bf[4];
#pragma unroll
        for (int ni = 0; ni < 4; ++ni) {
            const _Float16* pb = Bs + (n_off + ni * 16 + lidx) * LDT + lhalf * 8;
            half8_t blo = *(const half8_t*)pb;          // K = koff .. koff+7
            half8_t bhi = *(const half8_t*)(pb + 16);   // K = 16+koff .. 23+koff
            bf[ni] = __builtin_shufflevector(blo, bhi,
                     0, 1, 2, 3, 4, 5, 6, 7, 8, 9, 10, 11, 12, 13, 14, 15);
        }

#pragma unroll
        for (int mi = 0; mi < 2; ++mi)
#pragma unroll
            for (int ni = 0; ni < 4; ++ni)
                acc[mi][ni] = __builtin_amdgcn_wmma_f32_16x16x32_f16(
                    false, af[mi], false, bf[ni], (short)0, acc[mi][ni],
                    false, false);
        __syncthreads();
    }

    // ---- epilogue: bias (+ReLU), C/D layout: lane -> col, vgpr r -> row ----
#pragma unroll
    for (int mi = 0; mi < 2; ++mi) {
#pragma unroll
        for (int ni = 0; ni < 4; ++ni) {
            const int gn = n0 + n_off + ni * 16 + lidx;
            const float bv = bias[gn];
#pragma unroll
            for (int r = 0; r < 8; ++r) {
                const int gm = m0 + m_off + mi * 16 + r + lhalf * 8;
                float v = acc[mi][ni][r] + bv;
                if (relu) v = fmaxf(v, 0.f);
                C[(size_t)gm * N + gn] = v;
            }
        }
    }
}

// =====================================================================
// Token embedding: circular Conv1d(C_IN -> D, k=3), bias-free
// out[b,l,d] = sum_{c,k} x[b,(l+k-1)%L,c] * Wtok[d,c,k]
// =====================================================================
__global__ void token_embed_k(const float* __restrict__ x,
                              const float* __restrict__ Wtok,
                              float* __restrict__ out)
{
    int idx = blockIdx.x * blockDim.x + threadIdx.x;
    if (idx >= M_ * D_) return;
    const int d  = idx & (D_ - 1);
    const int bl = idx >> 9;
    const int l  = bl % L_;
    const int b  = bl / L_;
    float s = 0.f;
#pragma unroll
    for (int k = 0; k < 3; ++k) {
        int lt = l + k - 1;
        lt = (lt + L_) % L_;
        const float* xr = x + ((size_t)b * L_ + lt) * CIN_;
        const float* wr = Wtok + d * (CIN_ * 3) + k;
#pragma unroll
        for (int c = 0; c < CIN_; ++c) s += xr[c] * wr[c * 3];
    }
    out[idx] = s;
}

// =====================================================================
// Fused residual add + series_decomp (moving avg k=3, replicate pad):
// s(j) = x[b,j,d] + y[b,j,d];  out = s(l) - (s(l-1)+s(l)+s(l+1))/3
// =====================================================================
__global__ void add_decomp_k(const float* __restrict__ x,
                             const float* __restrict__ y,
                             float* __restrict__ out)
{
    int idx = blockIdx.x * blockDim.x + threadIdx.x;
    if (idx >= M_ * D_) return;
    const int d  = idx & (D_ - 1);
    const int bl = idx >> 9;
    const int l  = bl % L_;
    const int b  = bl / L_;
    const int lm = (l > 0) ? l - 1 : 0;
    const int lp = (l < L_ - 1) ? l + 1 : L_ - 1;
    const size_t base = (size_t)b * L_;
    const size_t om = (base + lm) * D_ + d;
    const size_t oc = (base + l)  * D_ + d;
    const size_t op = (base + lp) * D_ + d;
    const float sm = x[om] + y[om];
    const float sc = x[oc] + y[oc];
    const float sp = x[op] + y[op];
    out[idx] = sc - (sm + sc + sp) * (1.f / 3.f);
}

// =====================================================================
// AutoCorrelation (inference path), L=10, TOPK=6.
// One block per batch, one thread per channel d = h*DH+dh.
// corr[l] = sum_t q[t]*k[(t-l) mod L]; mean over 512 channels via ds_add;
// top-6 + softmax on thread 0; lag-gather of V through LDS.
// Output may safely alias the K buffer.
// =====================================================================
__global__ __launch_bounds__(512)
void autocorr_k(const float* __restrict__ Q, const float* __restrict__ K,
                const float* __restrict__ V, float* __restrict__ out)
{
    __shared__ float cm[L_];
    __shared__ float vsh[L_ * D_];
    __shared__ float sw[TOPK_];
    __shared__ int   sd[TOPK_];

    const int b = blockIdx.x;
    const int d = threadIdx.x;

    float q[L_], k[L_];
#pragma unroll
    for (int l = 0; l < L_; ++l) {
        const size_t o = ((size_t)b * L_ + l) * D_ + d;
        q[l] = Q[o];
        k[l] = K[o];
        vsh[l * D_ + d] = V[o];
    }
    if (d < L_) cm[d] = 0.f;
    __syncthreads();

#pragma unroll
    for (int l = 0; l < L_; ++l) {
        float c = 0.f;
#pragma unroll
        for (int t = 0; t < L_; ++t) c += q[t] * k[(t - l + L_) % L_];
        atomicAdd(&cm[l], c);   // ds_add_f32
    }
    __syncthreads();

    if (d == 0) {
        float w[TOPK_];
        int   id[TOPK_];
        for (int j = 0; j < TOPK_; ++j) {
            int bi = 0;
            float bv = cm[0];
            for (int i = 1; i < L_; ++i)
                if (cm[i] > bv) { bv = cm[i]; bi = i; }
            w[j]  = bv * (1.f / (float)D_);   // mean over H*DH channels
            id[j] = bi;
            cm[bi] = -3.0e38f;
        }
        float mx = w[0];
        for (int j = 1; j < TOPK_; ++j) mx = fmaxf(mx, w[j]);
        float ssum = 0.f;
        for (int j = 0; j < TOPK_; ++j) { w[j] = __expf(w[j] - mx); ssum += w[j]; }
        const float inv = 1.f / ssum;
        for (int j = 0; j < TOPK_; ++j) { sw[j] = w[j] * inv; sd[j] = id[j]; }
    }
    __syncthreads();

#pragma unroll
    for (int l = 0; l < L_; ++l) {
        float s = 0.f;
        for (int j = 0; j < TOPK_; ++j) {
            int li = l + sd[j];
            if (li >= L_) li -= L_;
            s += sw[j] * vsh[li * D_ + d];
        }
        out[((size_t)b * L_ + l) * D_ + d] = s;
    }
}

// =====================================================================
// Final: my_layernorm (LN over D, then subtract mean over L) + two tiny
// projections + sigmoid. One block per batch, one thread per channel d.
// =====================================================================
__device__ __forceinline__ float block_reduce_sum(float v, float* red)
{
    const int t = threadIdx.x;
    red[t] = v;
    __syncthreads();
#pragma unroll
    for (int s = 256; s > 0; s >>= 1) {
        if (t < s) red[t] += red[t + s];
        __syncthreads();
    }
    const float r = red[0];
    __syncthreads();
    return r;
}

__global__ __launch_bounds__(512)
void final_k(const float* __restrict__ X, const float* __restrict__ gamma,
             const float* __restrict__ beta, const float* __restrict__ W1,
             const float* __restrict__ b1, const float* __restrict__ W2,
             const float* __restrict__ b2, float* __restrict__ out)
{
    __shared__ float red[512];
    const int b = blockIdx.x;
    const int d = threadIdx.x;

    float xl[L_];
#pragma unroll
    for (int l = 0; l < L_; ++l)
        xl[l] = X[((size_t)b * L_ + l) * D_ + d];

    const float g  = gamma[d];
    const float be = beta[d];
    float accum = 0.f;
#pragma unroll
    for (int l = 0; l < L_; ++l) {
        const float mu  = block_reduce_sum(xl[l], red) * (1.f / (float)D_);
        const float df  = xl[l] - mu;
        const float var = block_reduce_sum(df * df, red) * (1.f / (float)D_);
        const float xh  = df * rsqrtf(var + 1e-5f) * g + be;
        xl[l] = xh;
        accum += xh;
    }
    const float md = accum * (1.f / (float)L_);
    const float w1 = W1[d];
    float t = 0.f;
#pragma unroll
    for (int l = 0; l < L_; ++l) t += W2[l] * ((xl[l] - md) * w1);
    const float S = block_reduce_sum(t, red);
    if (d == 0) {
        float sw2 = 0.f;
#pragma unroll
        for (int l = 0; l < L_; ++l) sw2 += W2[l];
        const float o = S + b1[0] * sw2 + b2[0];
        out[b] = 1.f / (1.f + __expf(-o));
    }
}

// =====================================================================
// Orchestration
// =====================================================================
extern "C" void kernel_launch(void* const* d_in, const int* in_sizes, int n_in,
                              void* d_out, int out_size, void* d_ws, size_t ws_size,
                              hipStream_t stream)
{
    (void)in_sizes; (void)n_in; (void)out_size; (void)ws_size;

    const float* x_enc = (const float*)d_in[0];
    const float* Wtok  = (const float*)d_in[1];
    const float* gamma = (const float*)d_in[38];
    const float* beta  = (const float*)d_in[39];
    const float* W1    = (const float*)d_in[40];
    const float* b1    = (const float*)d_in[41];
    const float* W2    = (const float*)d_in[42];
    const float* b2    = (const float*)d_in[43];

    const size_t MD = (size_t)M_ * D_;
    float* buf[5];
    for (int i = 0; i < 5; ++i) buf[i] = (float*)d_ws + (size_t)i * MD;

    const dim3 bElem(256);
    const dim3 gElem((M_ * D_ + 255) / 256);
    const dim3 gGemm(M_ / BM, D_ / BN);

    token_embed_k<<<gElem, bElem, 0, stream>>>(x_enc, Wtok, buf[0]);

    int xi = 0;
    for (int layer = 0; layer < 3; ++layer) {
        const int base = 2 + 12 * layer;
        const float* Wq  = (const float*)d_in[base + 0];
        const float* bq  = (const float*)d_in[base + 1];
        const float* Wk  = (const float*)d_in[base + 2];
        const float* bk  = (const float*)d_in[base + 3];
        const float* Wv  = (const float*)d_in[base + 4];
        const float* bv  = (const float*)d_in[base + 5];
        const float* Wo  = (const float*)d_in[base + 6];
        const float* bo  = (const float*)d_in[base + 7];
        const float* Wc1 = (const float*)d_in[base + 8];
        const float* bc1 = (const float*)d_in[base + 9];
        const float* Wc2 = (const float*)d_in[base + 10];
        const float* bc2 = (const float*)d_in[base + 11];

        float* X = buf[xi];
        float* q = buf[(xi + 1) % 5];
        float* k = buf[(xi + 2) % 5];
        float* v = buf[(xi + 3) % 5];
        float* t = buf[(xi + 4) % 5];

        gemm_wmma_f16<<<gGemm, bElem, 0, stream>>>(X, Wq, bq, q, 0);
        gemm_wmma_f16<<<gGemm, bElem, 0, stream>>>(X, Wk, bk, k, 0);
        gemm_wmma_f16<<<gGemm, bElem, 0, stream>>>(X, Wv, bv, v, 0);
        autocorr_k<<<B_, 512, 0, stream>>>(q, k, v, k);     // out aliases K (safe)
        gemm_wmma_f16<<<gGemm, bElem, 0, stream>>>(k, Wo, bo, t, 0);
        add_decomp_k<<<gElem, bElem, 0, stream>>>(X, t, q); // q = decomp(x + attn)
        gemm_wmma_f16<<<gGemm, bElem, 0, stream>>>(q, Wc1, bc1, k, 1); // ReLU
        gemm_wmma_f16<<<gGemm, bElem, 0, stream>>>(k, Wc2, bc2, t, 0);
        add_decomp_k<<<gElem, bElem, 0, stream>>>(q, t, v); // v = next layer input
        xi = (xi + 3) % 5;
    }

    final_k<<<B_, 512, 0, stream>>>(buf[xi], gamma, beta, W1, b1, W2, b2,
                                    (float*)d_out);
}
